// FeatureFlipFusion_57870389347084
// MI455X (gfx1250) — matrix operand order repaired and checked by reading, not compiled, not run on hardware
//
#include <hip/hip_runtime.h>
#include <hip/hip_bf16.h>
#include <math.h>

// ---------------------------------------------------------------------------
// FeatureFlipFusion for MI455X (gfx1250, wave32, WMMA).
// All GEMMs use v_wmma_f32_16x16x32_f16 (f16 in, f32 accumulate).
// A fragments: 2x global_load_b128 (16 contiguous K per lane half, same order as B).
// B fragments: 2x ds_load_b128 from transposed, pad-strided LDS tiles.
// ---------------------------------------------------------------------------

typedef _Float16 h16;
typedef __attribute__((ext_vector_type(16))) _Float16 v16h;
typedef __attribute__((ext_vector_type(8)))  _Float16 v8h;
typedef __attribute__((ext_vector_type(8)))  float    v8f;

static constexpr int B_  = 8;
static constexpr int C_  = 256;
static constexpr int H_  = 45;
static constexpr int W_  = 80;
static constexpr int HW_ = H_ * W_;          // 3600
static constexpr int CK_ = C_ * 9;           // 2304
static constexpr int NTOT_ = B_ * C_ * HW_;  // 7,372,800

__device__ __forceinline__ int iclampi(int v, int lo, int hi) {
  return v < lo ? lo : (v > hi ? hi : v);
}

__device__ __forceinline__ v8h ld8(const h16* p) { return *(const v8h*)p; }

__device__ __forceinline__ v16h cat8(v8h lo, v8h hi) {
  return __builtin_shufflevector(lo, hi, 0, 1, 2, 3, 4, 5, 6, 7,
                                 8, 9, 10, 11, 12, 13, 14, 15);
}

__device__ __forceinline__ v8f wmma_f16(v16h a, v16h b, v8f c) {
  return __builtin_amdgcn_wmma_f32_16x16x32_f16(false, a, false, b,
                                                (short)0, c, false, false);
}

// A fragment: lane (halfsel) holds 16 contiguous K at k0 + halfsel*16
// (matches the B fragment's contiguous K order).
__device__ __forceinline__ v16h load_a_frag(const h16* arow, int k0, int halfsel) {
  return cat8(ld8(arow + k0 + halfsel * 16),
              ld8(arow + k0 + halfsel * 16 + 8));
}

// B fragment from a transposed LDS tile row (K-contiguous per pixel):
// lanes 0-15 hold K base+0..15 of column bn, lanes 16-31 hold K base+16..31.
__device__ __forceinline__ v16h load_b_frag(const h16* brow) {
  return cat8(ld8(brow), ld8(brow + 8));
}

// ---------------------------------------------------------------------------
// Prep kernels
// ---------------------------------------------------------------------------
__global__ void cvt_f32_to_f16(const float* __restrict__ in, h16* __restrict__ out, int n) {
  int i = blockIdx.x * blockDim.x + threadIdx.x;
  if (i < n) out[i] = (h16)in[i];
}

// dcn_w [O=256][C=256][3][3] -> f16 [O][k] with k = c*9 + kk
__global__ void pack_dcn(const float* __restrict__ w, h16* __restrict__ out) {
  int i = blockIdx.x * blockDim.x + threadIdx.x;
  if (i >= C_ * CK_) return;
  int o = i / CK_, k = i % CK_;
  int c = k / 9, kk = k % 9;
  out[i] = (h16)w[(o * C_ + c) * 9 + kk];
}

// off_w [27][512][3][3] -> f16 [kk=9][o=32 padded][c=512]
__global__ void pack_offw(const float* __restrict__ w, h16* __restrict__ out) {
  int i = blockIdx.x * blockDim.x + threadIdx.x;
  if (i >= 9 * 32 * 512) return;
  int kk = i / (32 * 512);
  int r  = i % (32 * 512);
  int o = r / 512, c = r % 512;
  out[i] = (o < 27) ? (h16)w[(o * 512 + c) * 9 + kk] : (h16)0.f;
}

// ---------------------------------------------------------------------------
// Stage 1: y = conv1(x)  (per-batch 256x256 @ 256x3600 GEMM via WMMA)
// grid = B*225 blocks, 512 threads (16 waves, one 16-row M-tile each).
// Whole 256(K) x 16(N) x-tile staged once in LDS, transposed + padded.
// ---------------------------------------------------------------------------
static constexpr int C1_PAD = 264;  // 256 + 8 f16 -> 528 B rows (4-bank rotation)

__global__ __launch_bounds__(512)
void conv1_wmma(const h16* __restrict__ xh, const h16* __restrict__ w1h,
                float* __restrict__ y) {
  __shared__ h16 ldb[16 * C1_PAD];          // [px][k]
  int blk  = blockIdx.x;
  int b    = blk / 225;
  int pix0 = (blk % 225) * 16;
  int tid  = threadIdx.x;

  {  // cooperative fill: coalesced global reads (32B per 16 threads)
    int col  = tid & 15;
    int row0 = tid >> 4;                     // 0..31
#pragma unroll
    for (int rr = 0; rr < 8; ++rr) {
      int row = row0 + 32 * rr;              // channel
      ldb[col * C1_PAD + row] = xh[((size_t)(b * C_ + row)) * HW_ + pix0 + col];
    }
  }
  __syncthreads();

  int wave = tid >> 5, lane = tid & 31;
  int halfsel = lane >> 4;
  int am = lane & 15, bn = lane & 15;
  int m0 = wave * 16;
  const h16* arow = w1h + (m0 + am) * C_;
  const h16* brow_base = &ldb[bn * C1_PAD];

  v8f acc = {};
  for (int ks = 0; ks < C_ / 32; ++ks) {
    int k0 = ks * 32;
    v16h a  = load_a_frag(arow, k0, halfsel);
    v16h bm = load_b_frag(brow_base + k0 + halfsel * 16);
    acc = wmma_f16(a, bm, acc);
  }
#pragma unroll
  for (int r = 0; r < 8; ++r) {
    int o = m0 + r + 8 * halfsel;   // D: lanes 0-15 -> M=r, lanes 16-31 -> M=r+8
    y[((size_t)(b * C_ + o)) * HW_ + pix0 + bn] = acc[r];
  }
}

// ---------------------------------------------------------------------------
// BN stats: per-channel mean + rsqrt(var+eps). grid = C blocks, 256 threads.
// ---------------------------------------------------------------------------
__global__ __launch_bounds__(256)
void bn_stats(const float* __restrict__ y, float* __restrict__ stats) {
  int c = blockIdx.x;
  float s = 0.f, s2 = 0.f;
  for (int i = threadIdx.x; i < B_ * HW_; i += 256) {
    int b = i / HW_, p = i % HW_;
    float v = y[((size_t)(b * C_ + c)) * HW_ + p];
    s += v; s2 += v * v;
  }
  __shared__ float rs[256], rs2[256];
  rs[threadIdx.x] = s; rs2[threadIdx.x] = s2;
  __syncthreads();
  for (int st = 128; st > 0; st >>= 1) {
    if (threadIdx.x < st) {
      rs[threadIdx.x]  += rs[threadIdx.x + st];
      rs2[threadIdx.x] += rs2[threadIdx.x + st];
    }
    __syncthreads();
  }
  if (threadIdx.x == 0) {
    const float invN = 1.f / (float)(B_ * HW_);
    float m   = rs[0] * invN;
    float var = rs2[0] * invN - m * m;
    stats[c * 2 + 0] = m;
    stats[c * 2 + 1] = rsqrtf(var + 1e-5f);
  }
}

// xp = g*(y-m)*inv + b ; write f32 in place and f16 copy
__global__ void bn_apply(float* __restrict__ y, const float* __restrict__ stats,
                         const float* __restrict__ g, const float* __restrict__ bb,
                         h16* __restrict__ xph) {
  int i = blockIdx.x * blockDim.x + threadIdx.x;
  if (i >= NTOT_) return;
  int c = (i / HW_) % C_;
  float v = (y[i] - stats[c * 2]) * stats[c * 2 + 1] * g[c] + bb[c];
  y[i]   = v;
  xph[i] = (h16)v;
}

// ---------------------------------------------------------------------------
// Stage 2: 3x3 conv over concat(flip(x), xp) -> 27 ch (+bias), via WMMA.
// M padded to 32 (2 waves/block); grid = B*H*5 blocks, 64 threads.
// Each 32(K)x16(N) B tile staged cooperatively in LDS (transposed, padded).
// ---------------------------------------------------------------------------
static constexpr int OC_PAD = 40;  // 32 + 8 f16 -> 80 B rows

__global__ __launch_bounds__(64)
void offset_conv_wmma(const h16* __restrict__ xh, const h16* __restrict__ xph,
                      const h16* __restrict__ wofp, const float* __restrict__ offb,
                      float* __restrict__ offmask) {
  __shared__ h16 tile[16 * OC_PAD];          // [px][k]
  int blk = blockIdx.x;
  int b  = blk / (H_ * 5);
  int h  = (blk / 5) % H_;
  int w0 = (blk % 5) * 16;
  int tid  = threadIdx.x;
  int wave = tid >> 5, lane = tid & 31;
  int halfsel = lane >> 4;
  int am = lane & 15, bn = lane & 15;
  int m0 = wave * 16;
  int pxf = tid & 15;      // fill: pixel handled by this thread
  int kgf = tid >> 4;      // fill: k-group (0..3), 8 rows each

  v8f acc = {};
  for (int kk = 0; kk < 9; ++kk) {
    int dy = kk / 3 - 1, dx = kk % 3 - 1;
    int hh = h + dy;
    bool hok = (hh >= 0) && (hh < H_);
    int wwp = w0 + pxf + dx;
    bool pok = hok && (wwp >= 0) && (wwp < W_);
    for (int c0 = 0; c0 < 512; c0 += 32) {
      // ---- cooperative fill (flip realized by reversed W index)
#pragma unroll
      for (int j = 0; j < 8; ++j) {
        int kb = kgf * 8 + j;
        int ch = c0 + kb;
        h16 v = (h16)0.f;
        if (pok) {
          if (ch < C_)
            v = xh[((size_t)(b * C_ + ch)) * HW_ + hh * W_ + (W_ - 1 - wwp)];
          else
            v = xph[((size_t)(b * C_ + (ch - C_))) * HW_ + hh * W_ + wwp];
        }
        tile[pxf * OC_PAD + kb] = v;
      }
      __syncthreads();
      v16h a  = load_a_frag(wofp + (kk * 32 + m0 + am) * 512, c0, halfsel);
      v16h bm = load_b_frag(&tile[bn * OC_PAD + halfsel * 16]);
      acc = wmma_f16(a, bm, acc);
      __syncthreads();
    }
  }
#pragma unroll
  for (int r = 0; r < 8; ++r) {
    int o = m0 + r + 8 * halfsel;
    if (o < 27)
      offmask[(((size_t)(b * 27 + o)) * H_ + h) * W_ + w0 + bn] = acc[r] + offb[o];
  }
}

// ---------------------------------------------------------------------------
// Stage 3: deformable conv. One block per 16-pixel tile, 256 threads (8 waves).
// Bilinear-gather samp into LDS transposed [px][k] (padded rows), then
// dcn_w @ samp via WMMA. Split-K in 2 phases (channels 0-127 / 128-255).
// ---------------------------------------------------------------------------
static constexpr int SK_PAD = 1160;  // 1152 + 8 f16 -> 2320 B rows

__global__ __launch_bounds__(256)
void deform_wmma(const float* __restrict__ xp, const h16* __restrict__ dcnh,
                 const float* __restrict__ dcnb, const float* __restrict__ offmask,
                 float* __restrict__ dout) {
  __shared__ h16  samp[16 * SK_PAD];         // [px][k_local]
  __shared__ int   sh0[144], sh1[144], sw0[144], sw1[144];
  __shared__ float swt[144][4];

  int blk = blockIdx.x;
  int b  = blk / (H_ * 5);
  int h  = (blk / 5) % H_;
  int w0 = (blk % 5) * 16;
  int tid  = threadIdx.x;
  int wave = tid >> 5, lane = tid & 31;

  if (tid < 144) {                           // 9 taps x 16 pixels
    int kk = tid / 16, px = tid % 16;
    int w = w0 + px;
    size_t obase = (((size_t)(b * 27)) * H_ + h) * W_ + w;
    // offset.reshape(B,K,2,H,W): dy_k = ch 2k, dx_k = ch 2k+1, mask_k = ch 18+k
    float dy = offmask[obase + (size_t)(2 * kk) * HW_];
    float dx = offmask[obase + (size_t)(2 * kk + 1) * HW_];
    float mr = offmask[obase + (size_t)(18 + kk) * HW_];
    float msk = 1.f / (1.f + expf(-mr));
    float ph = dy + (float)(kk / 3 - 1) + (float)h;
    float pw = dx + (float)(kk % 3 - 1) + (float)w;
    float fh0 = floorf(ph), fw0 = floorf(pw);
    float lh = ph - fh0, lw = pw - fw0;
    int h0 = (int)fh0, w0i = (int)fw0;
    int h1 = h0 + 1, w1 = w0i + 1;
    bool h0v = (h0 >= 0) && (h0 < H_);
    bool h1v = (h1 >= 0) && (h1 < H_);
    bool w0v = (w0i >= 0) && (w0i < W_);
    bool w1v = (w1 >= 0) && (w1 < W_);
    swt[tid][0] = (h0v && w0v) ? (1.f - lh) * (1.f - lw) * msk : 0.f;
    swt[tid][1] = (h0v && w1v) ? (1.f - lh) * lw * msk : 0.f;
    swt[tid][2] = (h1v && w0v) ? lh * (1.f - lw) * msk : 0.f;
    swt[tid][3] = (h1v && w1v) ? lh * lw * msk : 0.f;
    sh0[tid] = iclampi(h0, 0, H_ - 1);
    sh1[tid] = iclampi(h1, 0, H_ - 1);
    sw0[tid] = iclampi(w0i, 0, W_ - 1);
    sw1[tid] = iclampi(w1, 0, W_ - 1);
  }
  __syncthreads();

  int halfsel = lane >> 4;
  int am = lane & 15, bn = lane & 15;
  v8f acc0 = {}, acc1 = {};

  for (int phase = 0; phase < 2; ++phase) {
    int cbase = phase * 128;
    // ---- gather phase: fill samp for channels [cbase, cbase+128)
    for (int i = wave; i < 144; i += 8) {
      int kk = i / 16, px = i % 16;
      int h0 = sh0[i], h1 = sh1[i], wA = sw0[i], wB = sw1[i];
      float t0 = swt[i][0], t1 = swt[i][1], t2 = swt[i][2], t3 = swt[i][3];
#pragma unroll
      for (int j = 0; j < 4; ++j) {
        int cl = lane + 32 * j;              // local channel 0..127
        const float* base = xp + ((size_t)(b * C_ + cbase + cl)) * HW_;
        float v = t0 * base[h0 * W_ + wA] + t1 * base[h0 * W_ + wB]
                + t2 * base[h1 * W_ + wA] + t3 * base[h1 * W_ + wB];
        samp[px * SK_PAD + cl * 9 + kk] = (h16)v;
      }
    }
    __syncthreads();
    // ---- WMMA phase over this half of K (36 steps of 32)
    for (int t = 0; t < 2; ++t) {
      int m0 = (wave * 2 + t) * 16;
      const h16* arow = dcnh + (size_t)(m0 + am) * CK_;
      const h16* brow = &samp[bn * SK_PAD];
      v8f acc = (t == 0) ? acc0 : acc1;
      for (int ks = 0; ks < 36; ++ks) {
        int klocal = ks * 32;
        int kglob  = phase * 1152 + klocal;
        // keep the A stream flowing (lowers to global_prefetch_b8)
        __builtin_prefetch(arow + kglob + 64, 0, 1);
        v16h a  = load_a_frag(arow, kglob, halfsel);
        v16h bm = load_b_frag(brow + klocal + halfsel * 16);
        acc = wmma_f16(a, bm, acc);
      }
      if (t == 0) acc0 = acc; else acc1 = acc;
    }
    __syncthreads();
  }

#pragma unroll
  for (int t = 0; t < 2; ++t) {
    int m0 = (wave * 2 + t) * 16;
    v8f acc = (t == 0) ? acc0 : acc1;
#pragma unroll
    for (int r = 0; r < 8; ++r) {
      int o = m0 + r + 8 * halfsel;
      dout[(((size_t)(b * C_ + o)) * H_ + h) * W_ + w0 + bn] = acc[r] + dcnb[o];
    }
  }
}

// ---------------------------------------------------------------------------
// Stage 4: out = relu(BN(d) + xp)
// ---------------------------------------------------------------------------
__global__ void final_fuse(const float* __restrict__ d, const float* __restrict__ stats,
                           const float* __restrict__ g, const float* __restrict__ bb,
                           const float* __restrict__ xp, float* __restrict__ out) {
  int i = blockIdx.x * blockDim.x + threadIdx.x;
  if (i >= NTOT_) return;
  int c = (i / HW_) % C_;
  float v = (d[i] - stats[c * 2]) * stats[c * 2 + 1] * g[c] + bb[c] + xp[i];
  out[i] = v > 0.f ? v : 0.f;
}

// ---------------------------------------------------------------------------
// Host launch
// ---------------------------------------------------------------------------
extern "C" void kernel_launch(void* const* d_in, const int* in_sizes, int n_in,
                              void* d_out, int out_size, void* d_ws, size_t ws_size,
                              hipStream_t stream) {
  const float* x       = (const float*)d_in[0];
  const float* conv1_w = (const float*)d_in[1];
  const float* bn1_g   = (const float*)d_in[2];
  const float* bn1_b   = (const float*)d_in[3];
  const float* off_w   = (const float*)d_in[4];
  const float* off_b   = (const float*)d_in[5];
  const float* dcn_w   = (const float*)d_in[6];
  const float* dcn_b   = (const float*)d_in[7];
  const float* norm_g  = (const float*)d_in[8];
  const float* norm_b  = (const float*)d_in[9];
  float* out = (float*)d_out;

  // Workspace carve-up (~94 MB total, 256B-aligned slices).
  char* p = (char*)d_ws;
  auto alloc = [&](size_t bytes) {
    void* r = (void*)p;
    p += (bytes + 255) & ~(size_t)255;
    return r;
  };
  h16*   xh      = (h16*)  alloc((size_t)NTOT_ * 2);          // x in f16
  float* y       = (float*)alloc((size_t)NTOT_ * 4);          // conv1 out -> xp f32
  h16*   xph     = (h16*)  alloc((size_t)NTOT_ * 2);          // xp in f16
  float* offmask = (float*)alloc((size_t)27 * B_ * HW_ * 4);  // conv-out + bias
  float* dbuf    = (float*)alloc((size_t)NTOT_ * 4);          // deform out
  h16*   w1h     = (h16*)  alloc((size_t)C_ * C_ * 2);
  h16*   dcnh    = (h16*)  alloc((size_t)C_ * CK_ * 2);
  h16*   wofp    = (h16*)  alloc((size_t)9 * 32 * 512 * 2);
  float* stats1  = (float*)alloc((size_t)C_ * 2 * 4);
  float* stats2  = (float*)alloc((size_t)C_ * 2 * 4);

  const int T = 256;
  // Prep: conversions / weight packing
  cvt_f32_to_f16<<<(NTOT_ + T - 1) / T, T, 0, stream>>>(x, xh, NTOT_);
  cvt_f32_to_f16<<<(C_ * C_ + T - 1) / T, T, 0, stream>>>(conv1_w, w1h, C_ * C_);
  pack_dcn<<<(C_ * CK_ + T - 1) / T, T, 0, stream>>>(dcn_w, dcnh);
  pack_offw<<<(9 * 32 * 512 + T - 1) / T, T, 0, stream>>>(off_w, wofp);

  // Stage 1: 1x1 conv + BN
  conv1_wmma<<<B_ * 225, 512, 0, stream>>>(xh, w1h, y);
  bn_stats<<<C_, 256, 0, stream>>>(y, stats1);
  bn_apply<<<(NTOT_ + T - 1) / T, T, 0, stream>>>(y, stats1, bn1_g, bn1_b, xph);

  // Stage 2: offset/mask conv
  offset_conv_wmma<<<B_ * H_ * 5, 64, 0, stream>>>(xh, xph, wofp, off_b, offmask);

  // Stage 3: deformable conv
  deform_wmma<<<B_ * H_ * 5, 256, 0, stream>>>(y, dcnh, dcn_b, offmask, dbuf);

  // Stage 4: BN + residual + relu
  bn_stats<<<C_, 256, 0, stream>>>(dbuf, stats2);
  final_fuse<<<(NTOT_ + T - 1) / T, T, 0, stream>>>(dbuf, stats2, norm_g, norm_b, y, out);
}
